// Single_Model_9062380995273
// MI455X (gfx1250) — compile-verified, matrix operand order checked by reading, not run on hardware
//
#include <hip/hip_runtime.h>
#include <hip/hip_bf16.h>

// ---------------------------------------------------------------------------
// Transformer block for MI455X (gfx1250, wave32, WMMA).
// B=32 T=1024 NDIM=1024 D=128 H=4 dh=32 FFN=512
// GEMMs use v_wmma_f32_16x16x32_f16; weights pre-packed in fragment order and
// staged to LDS via global_load_async_to_lds_b128 (+ s_wait_asynccnt).
// V is produced in [B,H,dh,T] layout so attention B-fragments are b128 loads.
// ---------------------------------------------------------------------------

typedef __attribute__((ext_vector_type(16))) _Float16 v16h;
typedef __attribute__((ext_vector_type(8)))  float    v8f;
typedef __attribute__((ext_vector_type(4)))  int      v4i;

union Frag16 {
    v16h     v;
    _Float16 h[16];
    unsigned u[8];
};
union Half4 {
    _Float16 h[4];
    unsigned long long u;
};

#if __has_builtin(__builtin_amdgcn_global_load_async_to_lds_b128)
#define HAVE_ASYNC_LDS 1
typedef __attribute__((address_space(1))) v4i* gas_b128_t;
typedef __attribute__((address_space(3))) v4i* las_b128_t;
#else
#define HAVE_ASYNC_LDS 0
#endif

// Per-lane WMMA fragment indexing (CDNA5 ISA 7.12.2, wave32):
//  A (16x32 f16): lanes 0-15 and 16-31 both hold M = lane&15.
//    VGPR j (j<4): K = kb + 2j, kb = (lane>=16 ? 8 : 0); VGPR j>=4: K += 16.
//  B (32x16 f16): symmetric — lane holds column N = lane&15, same K pattern.
//  C/D (16x16 f32, 8 VGPRs): lane holds N = lane&15, rows M = j + (lane>=16 ? 8 : 0).
__device__ __forceinline__ int frag_k(int j, int kb) {
    return ((j < 4) ? 0 : 16) + kb + ((j & 3) << 1);
}

// ---------------------------------------------------------------------------
// Weight repack: f32 row-major [K,N] -> f16 fragment order.
// Layout: Bp[((tn*ktiles + kt)*32 + lane)*16 + j]; one wave per (tn,kt) tile.
// ---------------------------------------------------------------------------
__global__ __launch_bounds__(128) void pack_b_kernel(
    const float* __restrict__ B, _Float16* __restrict__ Bp, int K, int N)
{
    const int wave = blockIdx.x * 4 + (threadIdx.x >> 5);
    const int ktiles = K >> 5, ntiles = N >> 4;
    if (wave >= ktiles * ntiles) return;
    const int tn = wave / ktiles, kt = wave - tn * ktiles;
    const int lane = threadIdx.x & 31;
    const int r = lane & 15, kb = (lane >> 4) << 3;
    const float* Bcol = B + tn * 16 + r;
    _Float16* dst = Bp + (((size_t)tn * ktiles + kt) * 32 + lane) * 16;
#pragma unroll
    for (int j = 0; j < 8; ++j) {
        const int k0 = kt * 32 + frag_k(j, kb);
        dst[2 * j]     = (_Float16)Bcol[(size_t)k0 * N];
        dst[2 * j + 1] = (_Float16)Bcol[(size_t)(k0 + 1) * N];
    }
}

// ---------------------------------------------------------------------------
// WMMA GEMM: out = act( A[M,K](f16) @ B[K,N](packed f16) + bias + residual ).
// Block = 128 threads = 4 waves computing a 64x16 strip (same N-tile).
// Packed B tile (K*16 halfs, <=32KB) staged once into LDS (async).
// Optional outVT stores f16 result transposed per-head: [B, H, dh, T].
// ---------------------------------------------------------------------------
__global__ __launch_bounds__(128) void wmma_gemm_kernel(
    const _Float16* __restrict__ A, const _Float16* __restrict__ Bp,
    const float* __restrict__ bias, const float* __restrict__ residual,
    float* __restrict__ outF32, _Float16* __restrict__ outF16,
    _Float16* __restrict__ outVT,
    int M, int N, int K, int relu)
{
    __shared__ _Float16 Bsh[16384];               // up to K=1024 -> 32 KB
    const int lane   = threadIdx.x & 31;
    const int wv     = threadIdx.x >> 5;
    const int ntiles = N >> 4;
    const int tn     = blockIdx.x % ntiles;
    const int tm     = (blockIdx.x / ntiles) * 4 + wv;
    const int r      = lane & 15;
    const int kb     = (lane >> 4) << 3;

    // Stage this N-tile's packed B (K*16 halfs = K*32 bytes) into LDS.
    const _Float16* bsrc = Bp + (size_t)tn * K * 16;
#if HAVE_ASYNC_LDS
    {
        const char* g = (const char*)bsrc;
        char* l = (char*)Bsh;
        const int total = K * 32;
        for (int o = threadIdx.x * 16; o < total; o += 128 * 16) {
            __builtin_amdgcn_global_load_async_to_lds_b128(
                (gas_b128_t)(g + o), (las_b128_t)(l + o), 0, 0);
        }
#if __has_builtin(__builtin_amdgcn_s_wait_asynccnt)
        __builtin_amdgcn_s_wait_asynccnt(0);
#else
        asm volatile("s_wait_asynccnt 0x0" ::: "memory");
#endif
    }
#else
    for (int o = threadIdx.x; o < K * 16; o += 128) Bsh[o] = bsrc[o];
#endif
    __syncthreads();

    const _Float16* Arow = A + (size_t)(tm * 16 + r) * K;
    v8f acc = {};
    for (int kk = 0; kk < K; kk += 32) {
        Frag16 a, b;
        b.v = *reinterpret_cast<const v16h*>(Bsh + (((kk >> 5) * 32 + lane) << 4));
#pragma unroll
        for (int j = 0; j < 8; ++j)
            a.u[j] = *reinterpret_cast<const unsigned*>(Arow + kk + frag_k(j, kb));
        __builtin_prefetch(Arow + kk + 64, 0, 1);   // global_prefetch_b8
        acc = __builtin_amdgcn_wmma_f32_16x16x32_f16(
            false, a.v, false, b.v, (short)0, acc, false, false);
    }

    const int n     = tn * 16 + r;
    const float bv  = bias ? bias[n] : 0.0f;
    const int mbase = tm * 16 + ((lane >> 4) << 3);
#pragma unroll
    for (int j = 0; j < 8; ++j) {
        const size_t idx = (size_t)(mbase + j) * N + n;
        float vv = acc[j] + bv;
        if (residual) vv += residual[idx];
        if (relu)     vv = fmaxf(vv, 0.0f);
        if (outF32)   outF32[idx] = vv;
        if (outF16)   outF16[idx] = (_Float16)vv;
        if (outVT) {  // [B, H, dh, T]; 8 consecutive tokens per lane -> wide store
            const int mrow = mbase + j;
            outVT[((size_t)((mrow >> 10) * 4 + (n >> 5)) * 32 + (n & 31)) * 1024
                  + (mrow & 1023)] = (_Float16)vv;
        }
    }
}

// ---------------------------------------------------------------------------
// Attention: one wave per (batch, head, 16-query tile). dh=32 == one WMMA K.
// Scores (f32, 64KB) + probabilities (f16, 32KB) in LDS; V read from the
// per-head transposed layout so every fragment is two b128 loads.
// ---------------------------------------------------------------------------
__global__ __launch_bounds__(32) void attn_kernel(
    const _Float16* __restrict__ Q, const _Float16* __restrict__ Km,
    const _Float16* __restrict__ VT, _Float16* __restrict__ O)
{
    __shared__ float    sc[16 * 1024];            // 64 KB
    __shared__ _Float16 at[16 * 1024];            // 32 KB
    const int wid  = blockIdx.x;                  // b*256 + h*64 + qt
    const int qt   = wid & 63;
    const int h    = (wid >> 6) & 3;
    const int bb   = wid >> 8;
    const int lane = threadIdx.x;
    const int r    = lane & 15;
    const int kb   = (lane >> 4) << 3;
    const int mb   = (lane >> 4) << 3;
    const size_t rowbase = (size_t)bb * 1024 * 128;   // [B,T,D] flat
    const int hoff = h * 32;

    Frag16 qa;
    {
        const _Float16* qrow = Q + rowbase + (size_t)(qt * 16 + r) * 128 + hoff;
#pragma unroll
        for (int j = 0; j < 8; ++j)
            qa.u[j] = *reinterpret_cast<const unsigned*>(qrow + frag_k(j, kb));
    }

    const float scale = 0.17677669529663687f;     // 1/sqrt(32)
    for (int kt = 0; kt < 64; ++kt) {
        Frag16 kf;
        const _Float16* krow = Km + rowbase + (size_t)(kt * 16 + r) * 128 + hoff;
#pragma unroll
        for (int j = 0; j < 8; ++j)
            kf.u[j] = *reinterpret_cast<const unsigned*>(krow + frag_k(j, kb));
        v8f s = {};
        s = __builtin_amdgcn_wmma_f32_16x16x32_f16(
            false, qa.v, false, kf.v, (short)0, s, false, false);
        const int col = kt * 16 + r;
#pragma unroll
        for (int j = 0; j < 8; ++j)
            sc[(mb + j) * 1024 + col] = s[j] * scale;
    }
    __syncthreads();

    // Softmax stats for row r (both half-waves need them for their fragments).
    const float4* rowp = reinterpret_cast<const float4*>(sc + r * 1024);
    float mx = -1e30f;
    for (int c = 0; c < 256; ++c) {
        const float4 sv = rowp[c];
        mx = fmaxf(mx, fmaxf(fmaxf(sv.x, sv.y), fmaxf(sv.z, sv.w)));
    }
    float sum = 0.0f;
    for (int c = 0; c < 256; ++c) {
        const float4 sv = rowp[c];
        sum += __expf(sv.x - mx) + __expf(sv.y - mx)
             + __expf(sv.z - mx) + __expf(sv.w - mx);
    }
    const float inv = 1.0f / sum;

    // Normalize -> f16 probabilities. Lane covers row r, half the columns.
    {
        const int cb = (lane >> 4) * 512;
        for (int c = cb; c < cb + 512; c += 4) {
            const float4 sv = *reinterpret_cast<const float4*>(sc + r * 1024 + c);
            Half4 t;
            t.h[0] = (_Float16)(__expf(sv.x - mx) * inv);
            t.h[1] = (_Float16)(__expf(sv.y - mx) * inv);
            t.h[2] = (_Float16)(__expf(sv.z - mx) * inv);
            t.h[3] = (_Float16)(__expf(sv.w - mx) * inv);
            *reinterpret_cast<unsigned long long*>(at + r * 1024 + c) = t.u;
        }
    }
    __syncthreads();

    // O = P @ V: A-fragments from LDS (b128), B-fragments from VT (b128).
    const _Float16* arow = at + r * 1024;
    const _Float16* v0p  = VT + ((size_t)(bb * 4 + h) * 32 + r) * 1024;
    const _Float16* v1p  = v0p + 16 * 1024;
    v8f acc0 = {}, acc1 = {};
    for (int kk = 0; kk < 1024; kk += 32) {
        Frag16 a, b0, b1;
#pragma unroll
        for (int j = 0; j < 8; ++j) {
            const int k0 = kk + frag_k(j, kb);
            a.u[j]  = *reinterpret_cast<const unsigned*>(arow + k0);
            b0.u[j] = *reinterpret_cast<const unsigned*>(v0p + k0);
            b1.u[j] = *reinterpret_cast<const unsigned*>(v1p + k0);
        }
        acc0 = __builtin_amdgcn_wmma_f32_16x16x32_f16(
            false, a.v, false, b0.v, (short)0, acc0, false, false);
        acc1 = __builtin_amdgcn_wmma_f32_16x16x32_f16(
            false, a.v, false, b1.v, (short)0, acc1, false, false);
    }
#pragma unroll
    for (int j = 0; j < 8; ++j) {
        const size_t row = (size_t)(qt * 16 + mb + j);
        O[rowbase + row * 128 + hoff + r]      = (_Float16)acc0[j];
        O[rowbase + row * 128 + hoff + 16 + r] = (_Float16)acc1[j];
    }
}

// ---------------------------------------------------------------------------
// LayerNorm over D=128, one wave per row, f16 output for the next GEMM.
// ---------------------------------------------------------------------------
__global__ __launch_bounds__(128) void ln_kernel(
    const float* __restrict__ x, const float* __restrict__ g,
    const float* __restrict__ b, _Float16* __restrict__ y, int rows)
{
    const int wave = blockIdx.x * 4 + (threadIdx.x >> 5);
    if (wave >= rows) return;
    const int lane = threadIdx.x & 31;
    const float* row = x + (size_t)wave * 128;
    float v0 = row[lane], v1 = row[lane + 32], v2 = row[lane + 64], v3 = row[lane + 96];
    float s = v0 + v1 + v2 + v3;
    for (int m = 16; m >= 1; m >>= 1) s += __shfl_xor(s, m, 32);
    const float mean = s * (1.0f / 128.0f);
    float d0 = v0 - mean, d1 = v1 - mean, d2 = v2 - mean, d3 = v3 - mean;
    float vs = d0 * d0 + d1 * d1 + d2 * d2 + d3 * d3;
    for (int m = 16; m >= 1; m >>= 1) vs += __shfl_xor(vs, m, 32);
    const float inv = rsqrtf(vs * (1.0f / 128.0f) + 1e-6f);
    _Float16* yr = y + (size_t)wave * 128;
    yr[lane]      = (_Float16)(g[lane]      * d0 * inv + b[lane]);
    yr[lane + 32] = (_Float16)(g[lane + 32] * d1 * inv + b[lane + 32]);
    yr[lane + 64] = (_Float16)(g[lane + 64] * d2 * inv + b[lane + 64]);
    yr[lane + 96] = (_Float16)(g[lane + 96] * d3 * inv + b[lane + 96]);
}

// ---------------------------------------------------------------------------
// logits[row] = x3[row,:] . fc_w + fc_b    (D=128, one wave per row)
// ---------------------------------------------------------------------------
__global__ __launch_bounds__(128) void logits_kernel(
    const float* __restrict__ x, const float* __restrict__ w,
    const float* __restrict__ bias, float* __restrict__ out, int rows)
{
    const int wave = blockIdx.x * 4 + (threadIdx.x >> 5);
    if (wave >= rows) return;
    const int lane = threadIdx.x & 31;
    const float* row = x + (size_t)wave * 128;
    float s = row[lane] * w[lane] + row[lane + 32] * w[lane + 32]
            + row[lane + 64] * w[lane + 64] + row[lane + 96] * w[lane + 96];
    for (int m = 16; m >= 1; m >>= 1) s += __shfl_xor(s, m, 32);
    if (lane == 0) out[wave] = s + bias[0];
}

// ---------------------------------------------------------------------------
// Per-batch variable-k top-k mean + sigmoid. kk = seq_len/16 + 1 (<= 65).
// ---------------------------------------------------------------------------
__global__ __launch_bounds__(32) void cls_kernel(
    const float* __restrict__ logits, const int* __restrict__ seq_len,
    float* __restrict__ out)
{
    __shared__ float s[1024];
    const int b = blockIdx.x, lane = threadIdx.x;
    const int L = seq_len[b];
    for (int i = lane; i < 1024; i += 32)
        s[i] = (i < L) ? logits[(size_t)b * 1024 + i] : -1e9f;
    __syncthreads();
    const int kk = L / 16 + 1;
    float total = 0.0f;
    for (int it = 0; it < kk; ++it) {
        float bm = -1e30f; int bi = 0;
        for (int i = lane; i < 1024; i += 32) {
            float v = s[i];
            if (v > bm) { bm = v; bi = i; }
        }
        for (int m = 16; m >= 1; m >>= 1) {
            float om = __shfl_xor(bm, m, 32);
            int   oi = __shfl_xor(bi, m, 32);
            if (om > bm || (om == bm && oi < bi)) { bm = om; bi = oi; }
        }
        total += bm;
        if (lane == 0) s[bi] = -1e30f;
        __syncthreads();
    }
    if (lane == 0) out[b] = 1.0f / (1.0f + __expf(-total / (float)kk));
}

// ---------------------------------------------------------------------------
// fp32 -> f16 cast, 4-wide (b128 in, b64 out), for the big tensor f.
// ---------------------------------------------------------------------------
__global__ void cvt_kernel(const float4* __restrict__ in,
                           _Float16* __restrict__ out, int n4)
{
    int i = blockIdx.x * blockDim.x + threadIdx.x;
    const int stride = gridDim.x * blockDim.x;
    for (; i < n4; i += stride) {
        const float4 sv = in[i];
        Half4 t;
        t.h[0] = (_Float16)sv.x; t.h[1] = (_Float16)sv.y;
        t.h[2] = (_Float16)sv.z; t.h[3] = (_Float16)sv.w;
        *reinterpret_cast<unsigned long long*>(out + (size_t)i * 4) = t.u;
    }
}

// ---------------------------------------------------------------------------
extern "C" void kernel_launch(void* const* d_in, const int* in_sizes, int n_in,
                              void* d_out, int out_size, void* d_ws, size_t ws_size,
                              hipStream_t stream)
{
    (void)in_sizes; (void)n_in; (void)out_size; (void)ws_size;
    const float* f    = (const float*)d_in[0];
    const int*   seq  = (const int*)d_in[1];
    const float* fcvw = (const float*)d_in[2];
    const float* fcvb = (const float*)d_in[3];
    const float* wq   = (const float*)d_in[4];  const float* bq = (const float*)d_in[5];
    const float* wk   = (const float*)d_in[6];  const float* bk = (const float*)d_in[7];
    const float* wv   = (const float*)d_in[8];  const float* bv = (const float*)d_in[9];
    const float* wo   = (const float*)d_in[10]; const float* bo = (const float*)d_in[11];
    const float* ln1g = (const float*)d_in[12]; const float* ln1b = (const float*)d_in[13];
    const float* ln2g = (const float*)d_in[14]; const float* ln2b = (const float*)d_in[15];
    const float* w1   = (const float*)d_in[16]; const float* b1 = (const float*)d_in[17];
    const float* w2   = (const float*)d_in[18]; const float* b2 = (const float*)d_in[19];
    const float* fcw  = (const float*)d_in[20]; const float* fcb = (const float*)d_in[21];
    float* out = (float*)d_out;

    const int Rows = 32 * 1024;   // B*T
    char* ws = (char*)d_ws;
    size_t off = 0;
    auto alloc = [&](size_t bytes) -> void* {
        void* p = ws + off;
        off += (bytes + 255) & ~(size_t)255;
        return p;
    };

    _Float16* fH    = (_Float16*)alloc((size_t)Rows * 1024 * 2);
    _Float16* fcvwP = (_Float16*)alloc((size_t)1024 * 128 * 2);
    _Float16* wqP   = (_Float16*)alloc((size_t)128 * 128 * 2);
    _Float16* wkP   = (_Float16*)alloc((size_t)128 * 128 * 2);
    _Float16* wvP   = (_Float16*)alloc((size_t)128 * 128 * 2);
    _Float16* woP   = (_Float16*)alloc((size_t)128 * 128 * 2);
    _Float16* w1P   = (_Float16*)alloc((size_t)128 * 512 * 2);
    _Float16* w2P   = (_Float16*)alloc((size_t)512 * 128 * 2);
    float*    x     = (float*)   alloc((size_t)Rows * 128 * 4);
    float*    x2    = (float*)   alloc((size_t)Rows * 128 * 4);
    float*    x3    = (float*)   alloc((size_t)Rows * 128 * 4);
    _Float16* yH    = (_Float16*)alloc((size_t)Rows * 128 * 2);
    _Float16* qH    = (_Float16*)alloc((size_t)Rows * 128 * 2);
    _Float16* kH    = (_Float16*)alloc((size_t)Rows * 128 * 2);
    _Float16* vT    = (_Float16*)alloc((size_t)Rows * 128 * 2);  // [B,H,dh,T]
    _Float16* oH    = (_Float16*)alloc((size_t)Rows * 128 * 2);
    _Float16* hH    = (_Float16*)alloc((size_t)Rows * 512 * 2);
    float*    lg    = (float*)   alloc((size_t)Rows * 4);

    // Activation cast + weight repack (cast fused into pack).
    cvt_kernel<<<2048, 256, 0, stream>>>((const float4*)f, fH, Rows * 1024 / 4);
    pack_b_kernel<<<64, 128, 0, stream>>>(fcvw, fcvwP, 1024, 128);
    pack_b_kernel<<<8,  128, 0, stream>>>(wq, wqP, 128, 128);
    pack_b_kernel<<<8,  128, 0, stream>>>(wk, wkP, 128, 128);
    pack_b_kernel<<<8,  128, 0, stream>>>(wv, wvP, 128, 128);
    pack_b_kernel<<<8,  128, 0, stream>>>(wo, woP, 128, 128);
    pack_b_kernel<<<32, 128, 0, stream>>>(w1, w1P, 128, 512);
    pack_b_kernel<<<32, 128, 0, stream>>>(w2, w2P, 512, 128);

    auto gemm = [&](const _Float16* A, const _Float16* Bp, const float* bias,
                    const float* res, float* o32, _Float16* o16, _Float16* ovt,
                    int M, int N, int K, int relu) {
        const int blocks = (M / 64) * (N / 16);   // 4 waves/block over 64x16
        wmma_gemm_kernel<<<blocks, 128, 0, stream>>>(A, Bp, bias, res,
                                                     o32, o16, ovt, M, N, K, relu);
    };

    // x = f @ fc_v_w + b
    gemm(fH, fcvwP, fcvb, nullptr, x, nullptr, nullptr, Rows, 128, 1024, 0);
    // y = LN1(x)
    ln_kernel<<<Rows / 4, 128, 0, stream>>>(x, ln1g, ln1b, yH, Rows);
    // q,k projections; v written transposed per-head
    gemm(yH, wqP, bq, nullptr, nullptr, qH, nullptr, Rows, 128, 128, 0);
    gemm(yH, wkP, bk, nullptr, nullptr, kH, nullptr, Rows, 128, 128, 0);
    gemm(yH, wvP, bv, nullptr, nullptr, nullptr, vT, Rows, 128, 128, 0);
    // attention (B*H*T/16 = 8192 waves)
    attn_kernel<<<8192, 32, 0, stream>>>(qH, kH, vT, oH);
    // x2 = x + o @ wo + bo
    gemm(oH, woP, bo, x, x2, nullptr, nullptr, Rows, 128, 128, 0);
    // y = LN2(x2)
    ln_kernel<<<Rows / 4, 128, 0, stream>>>(x2, ln2g, ln2b, yH, Rows);
    // h = relu(y @ w1 + b1)
    gemm(yH, w1P, b1, nullptr, nullptr, hH, nullptr, Rows, 512, 128, 1);
    // x3 = x2 + h @ w2 + b2
    gemm(hH, w2P, b2, x2, x3, nullptr, nullptr, Rows, 128, 512, 0);
    // logits + classification head
    logits_kernel<<<Rows / 4, 128, 0, stream>>>(x3, fcw, fcb, lg, Rows);
    cls_kernel<<<32, 32, 0, stream>>>(lg, seq, out);
}
